// AttentionBasedRouter_8555574854212
// MI455X (gfx1250) — matrix operand order repaired and checked by reading, not compiled, not run on hardware
//
#include <hip/hip_runtime.h>

#define H     1024
#define HF    512        // H/2
#define NE    8
#define NTOK  32768      // B*S
#define KSLAB 32         // K rows per slab
#define NSLAB (H / KSLAB)
#define SLABDW 8192      // 512 cols * 16 kpairs (dwords) = 32KB per slab

typedef __bf16 bf16x16 __attribute__((ext_vector_type(16)));
typedef float  f32x8   __attribute__((ext_vector_type(8)));
typedef unsigned int u32x4 __attribute__((ext_vector_type(4)));
typedef int          i32x4 __attribute__((ext_vector_type(4)));
typedef int          i32x8 __attribute__((ext_vector_type(8)));

union AccFrag { f32x8 v; float f[8]; };
union BfFrag  { bf16x16 v; unsigned int u[8]; uint4 q[2]; };

#if __has_builtin(__builtin_amdgcn_tensor_load_to_lds) && __has_builtin(__builtin_amdgcn_s_wait_tensorcnt)
#define HAVE_TDM 1
#else
#define HAVE_TDM 0
#endif

// round-to-nearest-even f32->bf16, packed pair (lo=a, hi=b)
__device__ __forceinline__ unsigned int pack_bf16(float a, float b) {
    unsigned int ua = __builtin_bit_cast(unsigned int, a);
    unsigned int ub = __builtin_bit_cast(unsigned int, b);
    ua += 0x7FFFu + ((ua >> 16) & 1u);
    ub += 0x7FFFu + ((ub >> 16) & 1u);
    return (ua >> 16) | (ub & 0xFFFF0000u);
}

#if HAVE_TDM
// 1-D TDM copy of `ndw` dwords (ndw < 65536) from global to LDS.
// 6-arg builtin form (amdgpu-toolchain / therock-10.0 headers):
//   (uint32x4 g0, int32x8 g1, int32x4 g2, int32x4 g3, int32x8 pad, i32 cpol)
__device__ __forceinline__ void tdm_load_dwords(unsigned lds_addr, const void* gsrc,
                                                unsigned ndw) {
    unsigned long long ga = (unsigned long long)(uintptr_t)gsrc;
    u32x4 g0;
    g0[0] = 1u;                                      // count=1 valid descriptor
    g0[1] = lds_addr;                                // lds_addr [63:32]
    g0[2] = (unsigned)ga;                            // global_addr[31:0]
    g0[3] = (unsigned)(ga >> 32) | 0x80000000u;      // global_addr[56:32], type=2
    i32x8 g1;
    g1[0] = 0x00020000;                              // data_size=2 (4B elements)
    g1[1] = (int)(ndw << 16);                        // tensor_dim0[15:0]  @63:48
    g1[2] = (int)((ndw >> 16) & 0xFFFFu) | (1 << 16);// tensor_dim0[31:16], tensor_dim1=1
    g1[3] = (int)(ndw << 16);                        // tile_dim0 @127:112
    g1[4] = 0;                                       // tile_dim1/2 unused
    g1[5] = (int)ndw;                                // tensor_dim0_stride[31:0]
    g1[6] = 0;
    g1[7] = 0;
    i32x4 z4 = {0, 0, 0, 0};
    i32x8 z8 = {0, 0, 0, 0, 0, 0, 0, 0};
    __builtin_amdgcn_tensor_load_to_lds(g0, g1, z4, z4, z8, 0);
}
#endif

__global__ void zero_cnt(int* __restrict__ cnt) {
    if (threadIdx.x < NE) cnt[threadIdx.x] = 0;
}

// X [*, 2k] f32 -> bf16 pair dwords (pairs adjacent in memory along last dim)
__global__ void __launch_bounds__(256)
pack_x(const float* __restrict__ src, unsigned* __restrict__ dst, int ndw) {
    const size_t i = ((size_t)blockIdx.x * 256 + threadIdx.x) * 4;
    if ((int)i + 3 < ndw) {
        float4 a = *(const float4*)(src + 2 * i);
        float4 b = *(const float4*)(src + 2 * i + 4);
        uint4 o;
        o.x = pack_bf16(a.x, a.y); o.y = pack_bf16(a.z, a.w);
        o.z = pack_bf16(b.x, b.y); o.w = pack_bf16(b.z, b.w);
        *(uint4*)(dst + i) = o;
    }
}

// W [K][N] f32 -> slab-major WMMA-B layout: dst[(s*N + n)*16 + p] =
// pack(W[32s+2p][n], W[32s+2p+1][n]);  one k-slab x 512-col block is contiguous.
__global__ void __launch_bounds__(256)
pack_w(const float* __restrict__ src, unsigned* __restrict__ dst, int N, int nmask,
       int total) {
    const int idx = blockIdx.x * 256 + threadIdx.x;   // (s, n)
    if (idx >= total) return;
    const int n = idx & nmask;
    const int s = idx / N;
    const float* base = src + (size_t)s * 32 * N + n;
    unsigned* d = dst + (size_t)idx * 16;
#pragma unroll
    for (int p = 0; p < 16; ++p)
        d[p] = pack_bf16(base[(size_t)(2 * p) * N], base[(size_t)(2 * p + 1) * N]);
}

// ---------------------------------------------------------------------------
// Router: 16 tokens/block. h = relu(Xbf @ W1bf + b1) via WMMA (TDM-staged,
// double buffered), logits = h @ W2 + b2, argmax -> expert bucket.
// ---------------------------------------------------------------------------
__global__ void __launch_bounds__(128)
route_kernel(const unsigned* __restrict__ Xp, const unsigned* __restrict__ W1t,
             const float* __restrict__ b1, const float* __restrict__ W2,
             const float* __restrict__ b2, int* __restrict__ cnt,
             int* __restrict__ idxl)
{
    __shared__ union { unsigned b[2][SLABDW]; float h[16 * HF]; } lsB;  // 64KB
    __shared__ float ldsW2[HF * NE];
    __shared__ float ldsLogit[16 * NE];

    const int tid  = threadIdx.x;
    const int wave = tid >> 5;
    const int lane = tid & 31;
    const int half = lane >> 4;
    const int l15  = lane & 15;
    const int tok0 = blockIdx.x * 16;
    const int colbase = wave * 128;

    for (int i = tid; i < HF * NE; i += 128) ldsW2[i] = W2[i];

    AccFrag acc[8];
#pragma unroll
    for (int t = 0; t < 8; ++t) acc[t].v = (f32x8){0.f,0.f,0.f,0.f,0.f,0.f,0.f,0.f};

    const unsigned* xrow = Xp + (size_t)(tok0 + l15) * HF;  // 512 pair-dwords/row

    auto compute_slab = [&](int s, const unsigned* bb) {
        const int kp0 = s * 16;
        BfFrag af;
        af.q[0] = *(const uint4*)(xrow + kp0 + 4 * half);
        af.q[1] = *(const uint4*)(xrow + kp0 + 4 * half + 8);
#pragma unroll
        for (int t = 0; t < 8; ++t) {
            BfFrag bf;
            const unsigned* bp = bb + (colbase + t * 16 + l15) * 16 + 4 * half;
            bf.q[0] = *(const uint4*)bp;
            bf.q[1] = *(const uint4*)(bp + 8);
            acc[t].v = __builtin_amdgcn_wmma_f32_16x16x32_bf16(
                false, af.v, false, bf.v, (short)0, acc[t].v, false, false);
        }
    };

#if HAVE_TDM
    const unsigned lds0 = (unsigned)(uintptr_t)(void*)&lsB.b[0][0];
    if (wave == 0) tdm_load_dwords(lds0, W1t, SLABDW);
    for (int s = 0; s < NSLAB; ++s) {
        const int buf = s & 1;
        if (s + 1 < NSLAB) {
            if (wave == 0)
                tdm_load_dwords(lds0 + (buf ^ 1) * (SLABDW * 4),
                                W1t + (size_t)(s + 1) * SLABDW, SLABDW);
            __builtin_amdgcn_s_wait_tensorcnt(1);   // slab s landed; s+1 in flight
        } else {
            __builtin_amdgcn_s_wait_tensorcnt(0);
        }
        __syncthreads();
        compute_slab(s, &lsB.b[buf][0]);
        __syncthreads();
    }
#else
    for (int s = 0; s < NSLAB; ++s) {
        __syncthreads();
        const unsigned* gsrc = W1t + (size_t)s * SLABDW;
        for (int i = tid * 4; i < SLABDW; i += 512)
            *(uint4*)&lsB.b[0][i] = *(const uint4*)(gsrc + i);
        __syncthreads();
        compute_slab(s, &lsB.b[0][0]);
        __syncthreads();
    }
#endif

    __syncthreads();
#pragma unroll
    for (int t = 0; t < 8; ++t) {
        const int col = colbase + t * 16 + l15;
        const float bb = b1[col];
#pragma unroll
        for (int r = 0; r < 8; ++r) {
            const int m = r + half * 8;
            float v = acc[t].f[r] + bb;
            lsB.h[m * HF + col] = v > 0.f ? v : 0.f;
        }
    }
    __syncthreads();

    {   // logits: thread = (token = tid>>3, e = tid&7)
        const int mtok = tid >> 3;
        const int e    = tid & 7;
        float a = b2[e];
        const float* hr = &lsB.h[mtok * HF];
        for (int c = 0; c < HF; ++c) a += hr[c] * ldsW2[c * NE + e];
        ldsLogit[mtok * NE + e] = a;
    }
    __syncthreads();

    if ((tid & 7) == 0) {
        const int mtok = tid >> 3;
        const float* lg = &ldsLogit[mtok * NE];
        int best = 0;
        for (int e = 1; e < NE; ++e) if (lg[e] > lg[best]) best = e;  // first-max
        const int slot = atomicAdd(&cnt[best], 1);
        idxl[best * NTOK + slot] = tok0 + mtok;
    }
}

// ---------------------------------------------------------------------------
// Expert GEMM: (expert, token-tile, col-block); gather 16 tokens, TDM-stage
// expert_W slabs, bf16 WMMA with f32 accumulate, scatter f32 rows + bias.
// ---------------------------------------------------------------------------
__global__ void __launch_bounds__(128)
expert_kernel(const unsigned* __restrict__ Xp, const unsigned* __restrict__ eWt,
              const float* __restrict__ eB, const int* __restrict__ cnt,
              const int* __restrict__ idxl, float* __restrict__ out)
{
    __shared__ unsigned ldsB[2][SLABDW];   // 64KB
    __shared__ int ldsTok[16];

    const int e     = blockIdx.y;
    const int cb    = blockIdx.z;          // 512-col block
    const int cnt_e = cnt[e];
    const int base  = blockIdx.x * 16;
    if (base >= cnt_e) return;

    const int tid  = threadIdx.x;
    const int wave = tid >> 5;
    const int lane = tid & 31;
    const int half = lane >> 4;
    const int l15  = lane & 15;
    const int colbase = wave * 128;

    if (tid < 16) {
        const int i = base + tid;
        ldsTok[tid] = idxl[e * NTOK + (i < cnt_e ? i : base)];
    }
    __syncthreads();

    const unsigned* xrow = Xp + (size_t)ldsTok[l15] * HF;
    // slab-major eWt: slab (e*32+s), cols [cb*512, cb*512+512) is contiguous
    const unsigned* Wbase = eWt + ((size_t)e * NSLAB * H + (size_t)cb * HF) * 16;

    AccFrag acc[8];
#pragma unroll
    for (int t = 0; t < 8; ++t) acc[t].v = (f32x8){0.f,0.f,0.f,0.f,0.f,0.f,0.f,0.f};

    auto compute_slab = [&](int s, const unsigned* bb) {
        const int kp0 = s * 16;
        BfFrag af;
        af.q[0] = *(const uint4*)(xrow + kp0 + 4 * half);
        af.q[1] = *(const uint4*)(xrow + kp0 + 4 * half + 8);
#pragma unroll
        for (int t = 0; t < 8; ++t) {
            BfFrag bf;
            const unsigned* bp = bb + (colbase + t * 16 + l15) * 16 + 4 * half;
            bf.q[0] = *(const uint4*)bp;
            bf.q[1] = *(const uint4*)(bp + 8);
            acc[t].v = __builtin_amdgcn_wmma_f32_16x16x32_bf16(
                false, af.v, false, bf.v, (short)0, acc[t].v, false, false);
        }
    };

#if HAVE_TDM
    const unsigned lds0 = (unsigned)(uintptr_t)(void*)&ldsB[0][0];
    if (wave == 0) tdm_load_dwords(lds0, Wbase, SLABDW);
    for (int s = 0; s < NSLAB; ++s) {
        const int buf = s & 1;
        if (s + 1 < NSLAB) {
            if (wave == 0)
                tdm_load_dwords(lds0 + (buf ^ 1) * (SLABDW * 4),
                                Wbase + (size_t)(s + 1) * (H * 16), SLABDW);
            __builtin_amdgcn_s_wait_tensorcnt(1);
        } else {
            __builtin_amdgcn_s_wait_tensorcnt(0);
        }
        __syncthreads();
        compute_slab(s, &ldsB[buf][0]);
        __syncthreads();
    }
#else
    for (int s = 0; s < NSLAB; ++s) {
        __syncthreads();
        const unsigned* gsrc = Wbase + (size_t)s * (H * 16);
        for (int i = tid * 4; i < SLABDW; i += 512)
            *(uint4*)&ldsB[0][i] = *(const uint4*)(gsrc + i);
        __syncthreads();
        compute_slab(s, &ldsB[0][0]);
        __syncthreads();
    }
#endif

#pragma unroll
    for (int t = 0; t < 8; ++t) {
        const int col = cb * HF + colbase + t * 16 + l15;
        const float bb = eB[e * H + col];
#pragma unroll
        for (int r = 0; r < 8; ++r) {
            const int m = r + half * 8;
            if (base + m < cnt_e)
                out[(size_t)ldsTok[m] * H + col] = acc[t].f[r] + bb;
        }
    }
}

extern "C" void kernel_launch(void* const* d_in, const int* in_sizes, int n_in,
                              void* d_out, int out_size, void* d_ws, size_t ws_size,
                              hipStream_t stream) {
    const float* X  = (const float*)d_in[0];
    const float* W1 = (const float*)d_in[1];
    const float* b1 = (const float*)d_in[2];
    const float* W2 = (const float*)d_in[3];
    const float* b2 = (const float*)d_in[4];
    const float* eW = (const float*)d_in[5];
    const float* eB = (const float*)d_in[6];
    float* out = (float*)d_out;

    unsigned char* ws = (unsigned char*)d_ws;
    int* cnt        = (int*)ws;                                   // 32 B
    int* idxl       = (int*)(ws + 1024);                          // 1 MB
    unsigned* Xp    = (unsigned*)(ws + (2u << 20));               // 64 MB
    unsigned* W1t   = (unsigned*)(ws + (2u << 20) + (64u << 20)); // 1 MB
    unsigned* eWt   = (unsigned*)(ws + (2u << 20) + (65u << 20)); // 16 MB

    const int xdw = NTOK * HF;                       // 16.8M pair-dwords
    pack_x<<<(xdw / 4 + 255) / 256, 256, 0, stream>>>(X, Xp, xdw);
    pack_w<<<(32 * HF + 255) / 256, 256, 0, stream>>>(W1, W1t, HF, HF - 1, 32 * HF);
    pack_w<<<(256 * H + 255) / 256, 256, 0, stream>>>(eW, eWt, H, H - 1, 256 * H);
    zero_cnt<<<1, 32, 0, stream>>>(cnt);
    route_kernel<<<NTOK / 16, 128, 0, stream>>>(Xp, W1t, b1, W2, b2, cnt, idxl);
    expert_kernel<<<dim3(NTOK / 16, NE, 2), 128, 0, stream>>>(Xp, eWt, eB, cnt, idxl, out);
}
